// BaselineCrfTagger_50895362457655
// MI455X (gfx1250) — matrix operand order repaired and checked by reading, not compiled, not run on hardware
//
#include <hip/hip_runtime.h>
#include <hip/hip_bf16.h>
#include <hip/hip_fp16.h>

typedef __attribute__((ext_vector_type(16))) _Float16 v16h;
typedef __attribute__((ext_vector_type(8)))  _Float16 v8h;
typedef __attribute__((ext_vector_type(8)))  float    v8f;
typedef __attribute__((ext_vector_type(4)))  int      v4i;

union AFrag { v16h v; v8h h[2]; };

#define BIGF 10000.0f
#define NEGF (-1e30f)

// padded LDS strides (in halfs) to kill 16-way bank conflicts
#define XSTRIDE 520   // 1040 B  (260 dwords ≡ 4 mod 64)
#define BSTRIDE 40    //   80 B  ( 20 dwords ≡ 20 mod 64)
#define HSTRIDE 136   //  272 B  ( 68 dwords ≡ 4 mod 64)
#define W2STRIDE 520
#define BBUF (128 * BSTRIDE)         // halfs per ldsB buffer

#if defined(__gfx1250__) && __has_builtin(__builtin_amdgcn_global_load_async_to_lds_b128)
#define HAVE_ASYNC 1
#else
#define HAVE_ASYNC 0
#endif

#if HAVE_ASYNC
#if __has_builtin(__builtin_amdgcn_s_wait_asynccnt)
#define WAITASYNC(n) __builtin_amdgcn_s_wait_asynccnt(n)
#else
#define WAITASYNC(n) asm volatile("s_wait_asynccnt " #n ::: "memory")
#endif
#else
#define WAITASYNC(n)
#endif

// 16-byte global -> LDS copy (async DMA on CDNA5, sync fallback otherwise)
__device__ __forceinline__ void g2l16(const _Float16* g, _Float16* l) {
#if HAVE_ASYNC
  __builtin_amdgcn_global_load_async_to_lds_b128(
      (__attribute__((address_space(1))) v4i*)(g),
      (__attribute__((address_space(3))) v4i*)(l), 0, 0);
#else
  *(uint4*)l = *(const uint4*)g;
#endif
}

// ---------------------------------------------------------------------------
__global__ void cvt_f16_kernel(const float* __restrict__ src,
                               _Float16* __restrict__ dst, int n) {
  int i = blockIdx.x * blockDim.x + threadIdx.x;
  if (i < n) dst[i] = (_Float16)src[i];
}

__global__ void cvt_w2_kernel(const float* __restrict__ w2,
                              _Float16* __restrict__ dst) {
  int i = blockIdx.x * blockDim.x + threadIdx.x;  // 32*512
  if (i < 32 * 512) {
    int t = i >> 9, k = i & 511;
    dst[i] = (t < 24) ? (_Float16)w2[t * 512 + k] : (_Float16)0.0f;
  }
}

// ---------------------------------------------------------------------------
// Fused emissions kernel: one workgroup (8 waves) = one batch element.
// GEMM1 (M=128,N=512,K=512) in 4 N-phases, each phase's h tile fed straight
// into GEMM2 (K accumulation). w1 K-slices arrive via triple-buffered
// global_load_async_to_lds_b128 with prefetch distance 2.
// ---------------------------------------------------------------------------
__global__ void emis_kernel(const int* __restrict__ seq,
                            const _Float16* __restrict__ embh,   // [200][512]
                            const _Float16* __restrict__ w1h,    // [512][512]
                            const _Float16* __restrict__ w2h,    // [32][512]
                            const float* __restrict__ b1,        // [512]
                            const float* __restrict__ b2,        // [24]
                            float* __restrict__ em) {            // [131072][24]
  extern __shared__ char smem[];
  _Float16* ldsX  = (_Float16*)smem;                    // 128*520*2 = 133120 B
  _Float16* ldsB  = (_Float16*)(smem + 133120);         // 3*128*40*2=  30720 B
  _Float16* ldsH  = (_Float16*)(smem + 163840);         // 8*2176*2 =   34816 B
  _Float16* ldsW2 = (_Float16*)(smem + 198656);         // 32*520*2 =   33280 B
  int*      ldsTok = (int*)(smem + 231936);             //               512 B

  const int blk = blockIdx.x;
  const int tid = threadIdx.x;
  const int w   = tid >> 5;
  const int l   = tid & 31;
  const int lh  = l >> 4;
  const int ll  = l & 15;

  // ---- stage x-tile (gather emb rows) + w2 into LDS via async DMA ----
  {
    int row  = tid >> 1;
    int part = tid & 1;
    int tok  = seq[blk * 128 + row];
    if (part == 0) ldsTok[row] = tok;
    const _Float16* g = embh + (size_t)tok * 512 + part * 256;
    _Float16*       d = ldsX + row * XSTRIDE + part * 256;
#pragma unroll
    for (int i = 0; i < 32; ++i) g2l16(g + i * 8, d + i * 8);
  }
  {
    int row = tid >> 3, sub = tid & 7;        // 8 threads per w2 row
    const _Float16* g = w2h + (size_t)row * 512 + sub * 64;
    _Float16*       d = ldsW2 + row * W2STRIDE + sub * 64;
#pragma unroll
    for (int i = 0; i < 8; ++i) g2l16(g + i * 8, d + i * 8);
  }

  const int nl = tid >> 1, part = tid & 1;    // for w1 chunk staging
  auto issue_chunk = [&](int c) {
    int p = c >> 4, kk = c & 15;
    const _Float16* g = w1h + (size_t)(128 * p + nl) * 512 + 32 * kk + part * 16;
    _Float16*       d = ldsB + (c % 3) * BBUF + nl * BSTRIDE + part * 16;
    g2l16(g, d);
    g2l16(g + 8, d + 8);
  };
  issue_chunk(0);
  issue_chunk(1);

  v8f scoreF[2];
#pragma unroll
  for (int ts = 0; ts < 2; ++ts)
#pragma unroll
    for (int r = 0; r < 8; ++r) scoreF[ts][r] = 0.0f;

  v8f acc[8];
  for (int it = 0; it < 64; ++it) {           // 4 phases x 16 K-chunks
    const int p = it >> 4, kk = it & 15, buf = it % 3;

    if (kk == 0) {
#pragma unroll
      for (int ns = 0; ns < 8; ++ns)
#pragma unroll
        for (int r = 0; r < 8; ++r) acc[ns][r] = 0.0f;
    }

    // current chunk's async copies are the oldest <=2 outstanding groups
    if (it == 63) { WAITASYNC(0); } else { WAITASYNC(2); }
    __syncthreads();
    if (it + 2 < 64) issue_chunk(it + 2);     // prefetch distance 2

    AFrag a;
    {
      const _Float16* base = ldsX + (w * 16 + ll) * XSTRIDE + 32 * kk + 16 * lh;
      a.h[0] = *(const v8h*)(base);
      a.h[1] = *(const v8h*)(base + 8);
    }
#pragma unroll
    for (int g4 = 0; g4 < 2; ++g4) {          // 2 groups of 4 N-subtiles
      v16h bf[4];
#pragma unroll
      for (int j = 0; j < 4; ++j) {
        int ns = g4 * 4 + j;
        bf[j] = *(const v16h*)(ldsB + buf * BBUF + (16 * ns + ll) * BSTRIDE + 16 * lh);
      }
#pragma unroll
      for (int j = 0; j < 4; ++j) {
        int ns = g4 * 4 + j;
        acc[ns] = __builtin_amdgcn_wmma_f32_16x16x32_f16(
            false, a.v, false, bf[j], (short)0, acc[ns], false, false);
      }
    }

    if (kk == 15) {
      // ---- phase epilogue: h = relu(acc + b1) -> per-wave LDS scratch ----
#pragma unroll
      for (int ns = 0; ns < 8; ++ns) {
        float bb = b1[128 * p + 16 * ns + ll];
#pragma unroll
        for (int r = 0; r < 8; ++r) {
          float v = acc[ns][r] + bb;
          v = v > 0.0f ? v : 0.0f;
          int m = r + 8 * lh;
          ldsH[w * (16 * HSTRIDE) + m * HSTRIDE + 16 * ns + ll] = (_Float16)v;
        }
      }
      // ---- GEMM2 partial: scores += hPhase[16x128] * w2^T[128x32] ----
#pragma unroll
      for (int q = 0; q < 4; ++q) {
        AFrag a2;
        const _Float16* hb =
            ldsH + w * (16 * HSTRIDE) + ll * HSTRIDE + 32 * q + 16 * lh;
        a2.h[0] = *(const v8h*)(hb);
        a2.h[1] = *(const v8h*)(hb + 8);
        v16h bw[2];
#pragma unroll
        for (int ts = 0; ts < 2; ++ts)
          bw[ts] = *(const v16h*)(ldsW2 + (16 * ts + ll) * W2STRIDE + 128 * p +
                                  32 * q + 16 * lh);
#pragma unroll
        for (int ts = 0; ts < 2; ++ts)
          scoreF[ts] = __builtin_amdgcn_wmma_f32_16x16x32_f16(
              false, a2.v, false, bw[ts], (short)0, scoreF[ts], false, false);
      }
    }
  }

  // ---- final: scores = relu(scoreF + b2) (+BIG at t==0 for PAD tokens) ----
  const int grow0 = blk * 128 + w * 16;
#pragma unroll
  for (int ts = 0; ts < 2; ++ts) {
    int t = 16 * ts + ll;
    if (t < 24) {
      float bb = b2[t];
#pragma unroll
      for (int r = 0; r < 8; ++r) {
        int m = r + 8 * lh;
        float v = scoreF[ts][r] + bb;
        v = v > 0.0f ? v : 0.0f;
        if (t == 0 && ldsTok[w * 16 + m] == 0) v += BIGF;
        em[(size_t)(grow0 + m) * 24 + t] = v;
      }
    }
  }
}

// ---------------------------------------------------------------------------
// CRF NLL: one wave32 per batch element. Lanes 0..23 = tags.
// ---------------------------------------------------------------------------
__global__ void crf_kernel(const float* __restrict__ em,
                           const int* __restrict__ labels,
                           const float* __restrict__ startT,
                           const float* __restrict__ endT,
                           const float* __restrict__ trans,
                           float* __restrict__ llh) {
  const int wid  = (blockIdx.x * blockDim.x + threadIdx.x) >> 5;
  const int lane = threadIdx.x & 31;
  const bool act = lane < 24;

  float tr[24];
#pragma unroll
  for (int i = 0; i < 24; ++i) tr[i] = act ? trans[i * 24 + lane] : 0.0f;

  const float* emb_ = em + (size_t)wid * 128 * 24;
  float alpha = act ? (startT[lane] + emb_[lane]) : NEGF;

  for (int s = 1; s < 128; ++s) {
    float emv = act ? emb_[s * 24 + lane] : 0.0f;
    float vbuf[24];
    float m = NEGF;
#pragma unroll
    for (int i = 0; i < 24; ++i) {
      float v = __shfl(alpha, i, 32) + tr[i];
      vbuf[i] = v;
      m = fmaxf(m, v);
    }
    float acc = 0.0f;
#pragma unroll
    for (int i = 0; i < 24; ++i) acc += __expf(vbuf[i] - m);
    float na = m + __logf(acc) + emv;
    alpha = act ? na : NEGF;
  }

  float val = act ? (alpha + endT[lane]) : NEGF;
  float m = val;
#pragma unroll
  for (int off = 16; off > 0; off >>= 1) m = fmaxf(m, __shfl_xor(m, off, 32));
  float e = act ? __expf(val - m) : 0.0f;
#pragma unroll
  for (int off = 16; off > 0; off >>= 1) e += __shfl_xor(e, off, 32);
  float denom = m + __logf(e);

  const int* lab = labels + wid * 128;
  float part = 0.0f;
  for (int s = lane; s < 128; s += 32) {
    int ls = lab[s];
    part += emb_[s * 24 + ls];
    if (s < 127) part += trans[ls * 24 + lab[s + 1]];
  }
#pragma unroll
  for (int off = 16; off > 0; off >>= 1) part += __shfl_xor(part, off, 32);

  if (lane == 0) {
    float num = part + startT[lab[0]] + endT[lab[127]];
    llh[wid] = num - denom;
  }
}

// ---------------------------------------------------------------------------
__global__ void reduce_kernel(const float* __restrict__ llh,
                              float* __restrict__ out) {
  __shared__ float sm[256];
  int t = threadIdx.x;
  float v = llh[t] + llh[t + 256] + llh[t + 512] + llh[t + 768];
  sm[t] = v;
  __syncthreads();
  for (int st = 128; st > 0; st >>= 1) {
    if (t < st) sm[t] += sm[t + st];
    __syncthreads();
  }
  if (t == 0) out[0] = -sm[0] / 1024.0f;
}

// ---------------------------------------------------------------------------
extern "C" void kernel_launch(void* const* d_in, const int* in_sizes, int n_in,
                              void* d_out, int out_size, void* d_ws, size_t ws_size,
                              hipStream_t stream) {
  const int*   seq    = (const int*)d_in[0];
  const int*   labels = (const int*)d_in[1];
  const float* emb = (const float*)d_in[3];
  const float* w1  = (const float*)d_in[4];
  const float* b1  = (const float*)d_in[5];
  const float* w2  = (const float*)d_in[6];
  const float* b2  = (const float*)d_in[7];
  const float* st  = (const float*)d_in[8];
  const float* en  = (const float*)d_in[9];
  const float* tr  = (const float*)d_in[10];

  char* ws = (char*)d_ws;
  _Float16* embh = (_Float16*)(ws);                       // 204800 B
  _Float16* w1h  = (_Float16*)(ws + 204800);              // 524288 B
  _Float16* w2h  = (_Float16*)(ws + 729088);              //  32768 B
  float*    emis = (float*)(ws + 761856);                 // 12582912 B
  float*    llh  = (float*)(ws + 13344768);               //   4096 B

  cvt_f16_kernel<<<(102400 + 255) / 256, 256, 0, stream>>>(emb, embh, 102400);
  cvt_f16_kernel<<<(262144 + 255) / 256, 256, 0, stream>>>(w1, w1h, 262144);
  cvt_w2_kernel<<<(16384 + 255) / 256, 256, 0, stream>>>(w2, w2h);

  const int shbytes = 232448;
  (void)hipFuncSetAttribute((const void*)emis_kernel,
                            hipFuncAttributeMaxDynamicSharedMemorySize, shbytes);
  emis_kernel<<<1024, 256, shbytes, stream>>>(seq, embh, w1h, w2h, b1, b2, emis);

  crf_kernel<<<256, 128, 0, stream>>>(emis, labels, st, en, tr, llh);
  reduce_kernel<<<1, 256, 0, stream>>>(llh, (float*)d_out);
}
